// Decoder_83554293777070
// MI455X (gfx1250) — matrix operand order repaired and checked by reading, not compiled, not run on hardware
//
#include <hip/hip_runtime.h>
#include <hip/hip_bf16.h>

// ---------------- problem constants ----------------
#define BATCH   64
#define TIN     60
#define TOUT    31
#define UNITS   512
#define EMBD    300
#define VOCAB   34004
#define KXH     1344          // 300 emb + 512 attn + 20 pad + 512 h  (multiple of 32)
#define NZ      2048          // 4*UNITS
#define NTILE_V 2126          // ceil(34004/16)

typedef __attribute__((ext_vector_type(16))) __bf16 v16bf;
typedef __attribute__((ext_vector_type(8)))  __bf16 v8bf;
typedef __attribute__((ext_vector_type(8)))  float  v8f;

#define WMMA_BF16(a, b, c) \
    __builtin_amdgcn_wmma_f32_16x16x32_bf16(false, (a), false, (b), (short)0, (c), false, false)

// ---------------- WMMA fragment helpers ----------------
// A fragment: 16x32 bf16, lane (m = lane&15, h = lane>>4) holds
// K = k0 + 8h + [0..7]  and  K = k0 + 16 + 8h + [0..7]   (ISA 7.12.2)
__device__ __forceinline__ v16bf load_a_frag(const __bf16* A, int lda,
                                             int m_base, int k0, int lane) {
    const int h = (lane >> 4) & 1;
    const int m = lane & 15;
    const __bf16* p = A + (size_t)(m_base + m) * lda + k0 + 8 * h;
    v8bf lo = *(const v8bf*)p;          // 16B load
    v8bf hi = *(const v8bf*)(p + 16);   // 16B load
    v16bf a;
#pragma unroll
    for (int i = 0; i < 8; ++i) { a[i] = lo[i]; a[8 + i] = hi[i]; }
    return a;
}

// B fragment: 32x16 bf16 from pre-transposed weight BT[nrow][k] (ldk = K).
// nrow is the (pre-clamped) N row for this lane; lane kh = lane>>4 selects
// K = k0 + 8*kh + [0..7] and K = k0 + 16 + 8*kh + [0..7].  Branch-free: tail lanes pass a clamped nrow.
__device__ __forceinline__ v16bf load_b_frag(const __bf16* BT, int ldk,
                                             int nrow, int k0, int lane) {
    const int kh = (lane >> 4) & 1;
    const __bf16* p = BT + (size_t)nrow * ldk + k0 + 8 * kh;
    v8bf lo = *(const v8bf*)p;
    v8bf hi = *(const v8bf*)(p + 16);
    v16bf b;
#pragma unroll
    for (int i = 0; i < 8; ++i) { b[i] = lo[i]; b[8 + i] = hi[i]; }
    return b;
}

// Full-M (64-row) column strip GEMM, software-pipelined double buffer:
// next k-step's fragments are loaded before the current wmma chain so the
// loads stay in flight behind the matrix ops (partial loadcnt waits).
__device__ __forceinline__ void gemm_strip(const __bf16* __restrict__ A, int lda,
                                           const __bf16* __restrict__ BT, int K,
                                           int nrow, int lane, v8f c[4]) {
    v16bf b  = load_b_frag(BT, K, nrow, 0, lane);
    v16bf a0 = load_a_frag(A, lda,  0, 0, lane);
    v16bf a1 = load_a_frag(A, lda, 16, 0, lane);
    v16bf a2 = load_a_frag(A, lda, 32, 0, lane);
    v16bf a3 = load_a_frag(A, lda, 48, 0, lane);
    for (int k0 = 32; k0 < K; k0 += 32) {
        v16bf bn  = load_b_frag(BT, K, nrow, k0, lane);
        v16bf an0 = load_a_frag(A, lda,  0, k0, lane);
        v16bf an1 = load_a_frag(A, lda, 16, k0, lane);
        v16bf an2 = load_a_frag(A, lda, 32, k0, lane);
        v16bf an3 = load_a_frag(A, lda, 48, k0, lane);
        c[0] = WMMA_BF16(a0, b, c[0]);
        c[1] = WMMA_BF16(a1, b, c[1]);
        c[2] = WMMA_BF16(a2, b, c[2]);
        c[3] = WMMA_BF16(a3, b, c[3]);
        b = bn; a0 = an0; a1 = an1; a2 = an2; a3 = an3;
    }
    c[0] = WMMA_BF16(a0, b, c[0]);
    c[1] = WMMA_BF16(a1, b, c[1]);
    c[2] = WMMA_BF16(a2, b, c[2]);
    c[3] = WMMA_BF16(a3, b, c[3]);
}

// ---------------- one-time setup kernels ----------------
// WfcT_bf[n][k] = bf16(Wfc[k][n])   (N-major so B frags are contiguous)
__global__ void k_wfcT(const float* __restrict__ Wfc, __bf16* __restrict__ WfcT) {
    size_t gid = (size_t)blockIdx.x * blockDim.x + threadIdx.x;
    if (gid >= (size_t)VOCAB * UNITS) return;
    int n = (int)(gid / UNITS);
    int k = (int)(gid % UNITS);
    WfcT[gid] = (__bf16)Wfc[(size_t)k * VOCAB + n];
}

// WkuT_bf[n][k]: k<812 -> Wk[k][n]; 812<=k<832 -> 0; else Uk[k-832][n]
__global__ void k_wkuT(const float* __restrict__ Wk, const float* __restrict__ Uk,
                       __bf16* __restrict__ WkuT) {
    size_t gid = (size_t)blockIdx.x * blockDim.x + threadIdx.x;
    if (gid >= (size_t)NZ * KXH) return;
    int n = (int)(gid / KXH);
    int k = (int)(gid % KXH);
    float v = 0.0f;
    if (k < EMBD + UNITS)            v = Wk[(size_t)k * NZ + n];
    else if (k >= EMBD + UNITS + 20) v = Uk[(size_t)(k - (EMBD + UNITS + 20)) * NZ + n];
    WkuT[gid] = (__bf16)v;
}

// keys[b,t,u] = sum_k memory[b,t,k] * Wm[k,u]
__global__ void k_keys(const float* __restrict__ mem, const float* __restrict__ Wm,
                       float* __restrict__ keys) {
    __shared__ float sm[UNITS];
    const int row = blockIdx.x;                    // b*TIN + t
    const float* mrow = mem + (size_t)row * UNITS;
    for (int k = threadIdx.x; k < UNITS; k += blockDim.x) sm[k] = mrow[k];
    __syncthreads();
    for (int u = threadIdx.x; u < UNITS; u += blockDim.x) {
        float acc = 0.0f;
        for (int k = 0; k < UNITS; ++k) acc += sm[k] * Wm[(size_t)k * UNITS + u];
        keys[(size_t)row * UNITS + u] = acc;
    }
}

// init: xh = 0 (attn0 = 0, pad = 0), then h-region <- bf16(h0); h,c fp32 state
__global__ void k_init(const float* __restrict__ h0, const float* __restrict__ c0,
                       __bf16* __restrict__ xh, float* __restrict__ h,
                       float* __restrict__ c) {
    int gid = blockIdx.x * blockDim.x + threadIdx.x;
    if (gid < BATCH * KXH) {
        int b = gid / KXH, j = gid % KXH;
        __bf16 v = (__bf16)0.0f;
        if (j >= EMBD + UNITS + 20) v = (__bf16)h0[(size_t)b * UNITS + (j - (EMBD + UNITS + 20))];
        xh[gid] = v;
    }
    if (gid < BATCH * UNITS) {
        h[gid] = h0[gid];
        c[gid] = c0[gid];
    }
}

// fill embedding region of xh for step t
__global__ void k_embed(const int* __restrict__ dec, const float* __restrict__ emb,
                        __bf16* __restrict__ xh, int t) {
    const int b = blockIdx.x;
    const int tok = dec[(size_t)b * TOUT + t];
    const float* erow = emb + (size_t)tok * EMBD;
    for (int j = threadIdx.x; j < EMBD; j += blockDim.x)
        xh[(size_t)b * KXH + j] = (__bf16)erow[j];
}

// ---------------- per-step kernels ----------------
// z = xh @ WkuT^T + bk  (M=64, N=2048, K=1344).
// One wave owns a full N-column strip: 4 M-tiles share each B fragment.
__global__ __launch_bounds__(256) void k_cell_gemm(const __bf16* __restrict__ xh,
                                                   const __bf16* __restrict__ WkuT,
                                                   const float* __restrict__ bk,
                                                   float* __restrict__ z) {
    const int lane  = threadIdx.x & 31;
    const int wave  = threadIdx.x >> 5;
    const int ntile = blockIdx.x * 8 + wave;       // 0..127
    const int col   = ntile * 16 + (lane & 15);
    const float bias = bk[col];                    // fold bias into C init

    v8f c[4];
#pragma unroll
    for (int m = 0; m < 4; ++m)
#pragma unroll
        for (int r = 0; r < 8; ++r) c[m][r] = bias;

    gemm_strip(xh, KXH, WkuT, KXH, col, lane, c);

    const int half = lane >> 4;
#pragma unroll
    for (int m = 0; m < 4; ++m)
#pragma unroll
        for (int r = 0; r < 8; ++r) {
            int row = m * 16 + r + 8 * half;
            z[(size_t)row * NZ + col] = c[m][r];
        }
}

// LSTM pointwise (bias already folded into z): gates -> h_new, c_new
__global__ void k_lstm(const float* __restrict__ z,
                       float* __restrict__ h, float* __restrict__ c,
                       __bf16* __restrict__ xh) {
    int gid = blockIdx.x * blockDim.x + threadIdx.x;
    if (gid >= BATCH * UNITS) return;
    int b = gid >> 9, j = gid & (UNITS - 1);
    const float* zr = z + (size_t)b * NZ;
    float zi = zr[j];
    float zf = zr[UNITS + j];
    float zg = zr[2 * UNITS + j];
    float zo = zr[3 * UNITS + j];
    float ig = 1.0f / (1.0f + expf(-zi));
    float fg = 1.0f / (1.0f + expf(-zf));
    float gg = tanhf(zg);
    float og = 1.0f / (1.0f + expf(-zo));
    float cn = fg * c[gid] + ig * gg;
    float hn = og * tanhf(cn);
    c[gid] = cn;
    h[gid] = hn;
    xh[(size_t)b * KXH + (EMBD + UNITS + 20) + j] = (__bf16)hn;
}

// fused Bahdanau attention + attention_layer, one block per batch row
__global__ __launch_bounds__(256) void k_attn(const float* __restrict__ h,
                                              const float* __restrict__ keys,
                                              const float* __restrict__ vatt,
                                              const float* __restrict__ mem,
                                              const float* __restrict__ Wq,
                                              const float* __restrict__ Wa,
                                              __bf16* __restrict__ attnA,
                                              __bf16* __restrict__ xh) {
    __shared__ float sh_h[UNITS];
    __shared__ float sh_q[UNITS];
    __shared__ float sh_c[UNITS];
    __shared__ float sh_s[64];
    const int b   = blockIdx.x;
    const int tid = threadIdx.x;
    const int lane = tid & 31, wave = tid >> 5;

    for (int u = tid; u < UNITS; u += blockDim.x) sh_h[u] = h[(size_t)b * UNITS + u];
    __syncthreads();

    // q = h @ Wq
    for (int u = tid; u < UNITS; u += blockDim.x) {
        float acc = 0.0f;
        for (int k = 0; k < UNITS; ++k) acc += sh_h[k] * Wq[(size_t)k * UNITS + u];
        sh_q[u] = acc;
    }
    __syncthreads();

    // scores: one wave per set of t's, lanes reduce over u
    const float* krow = keys + (size_t)b * TIN * UNITS;
    for (int t = wave; t < TIN; t += 8) {
        float p = 0.0f;
        for (int u = lane; u < UNITS; u += 32)
            p += vatt[u] * tanhf(krow[(size_t)t * UNITS + u] + sh_q[u]);
        for (int m = 16; m > 0; m >>= 1) p += __shfl_xor(p, m, 32);
        if (lane == 0) sh_s[t] = p;
    }
    __syncthreads();

    // softmax over TIN (tiny: serial in thread 0)
    if (tid == 0) {
        float mx = sh_s[0];
        for (int t = 1; t < TIN; ++t) mx = fmaxf(mx, sh_s[t]);
        float s = 0.0f;
        for (int t = 0; t < TIN; ++t) { float e = expf(sh_s[t] - mx); sh_s[t] = e; s += e; }
        float inv = 1.0f / s;
        for (int t = 0; t < TIN; ++t) sh_s[t] *= inv;
    }
    __syncthreads();

    // context = align @ memory
    const float* mrow = mem + (size_t)b * TIN * UNITS;
    for (int u = tid; u < UNITS; u += blockDim.x) {
        float acc = 0.0f;
        for (int t = 0; t < TIN; ++t) acc += sh_s[t] * mrow[(size_t)t * UNITS + u];
        sh_c[u] = acc;
    }
    __syncthreads();

    // attn = concat(h, context) @ Wa ; emit bf16 copies for next GEMMs
    for (int u = tid; u < UNITS; u += blockDim.x) {
        float acc = 0.0f;
        for (int k = 0; k < UNITS; ++k) acc += sh_h[k] * Wa[(size_t)k * UNITS + u];
        for (int k = 0; k < UNITS; ++k) acc += sh_c[k] * Wa[(size_t)(UNITS + k) * UNITS + u];
        __bf16 v = (__bf16)acc;
        attnA[(size_t)b * UNITS + u]     = v;        // A matrix for logits GEMM
        xh[(size_t)b * KXH + EMBD + u]   = v;        // recurrent input next step
    }
}

// logits = attnA @ WfcT^T + bfc -> d_out[:, t, :]  (M=64, N=34004, K=512)
// One wave per N-tile: B fragment loaded once, reused across all 4 M-tiles.
// Vocab tail handled branch-free by clamping the B row (column-local in C).
__global__ __launch_bounds__(256) void k_logits(const __bf16* __restrict__ attnA,
                                                const __bf16* __restrict__ WfcT,
                                                const float* __restrict__ bfc,
                                                float* __restrict__ out, int t) {
    const int lane  = threadIdx.x & 31;
    const int wave  = threadIdx.x >> 5;
    const int ntile = blockIdx.x * 8 + wave;
    if (ntile >= NTILE_V) return;
    const int col   = ntile * 16 + (lane & 15);
    const bool valid = col < VOCAB;
    const int ncl   = valid ? col : (VOCAB - 1);   // clamped B row for tail lanes
    const float bias = valid ? bfc[col] : 0.0f;    // fold bias into C init

    v8f c[4];
#pragma unroll
    for (int m = 0; m < 4; ++m)
#pragma unroll
        for (int r = 0; r < 8; ++r) c[m][r] = bias;

    gemm_strip(attnA, UNITS, WfcT, UNITS, ncl, lane, c);

    if (!valid) return;
    const int half = lane >> 4;
#pragma unroll
    for (int m = 0; m < 4; ++m)
#pragma unroll
        for (int r = 0; r < 8; ++r) {
            int b = m * 16 + r + 8 * half;
            out[((size_t)b * TOUT + t) * VOCAB + col] = c[m][r];
        }
}

// ---------------- host launch ----------------
static inline size_t align256(size_t x) { return (x + 255) & ~(size_t)255; }

extern "C" void kernel_launch(void* const* d_in, const int* in_sizes, int n_in,
                              void* d_out, int out_size, void* d_ws, size_t ws_size,
                              hipStream_t stream) {
    const int*   dec   = (const int*)  d_in[0];
    const float* h0    = (const float*)d_in[1];
    const float* c0    = (const float*)d_in[2];
    const float* mem   = (const float*)d_in[3];
    const float* emb   = (const float*)d_in[4];
    const float* Wk    = (const float*)d_in[5];
    const float* Uk    = (const float*)d_in[6];
    const float* bk    = (const float*)d_in[7];
    const float* Wm    = (const float*)d_in[8];
    const float* Wq    = (const float*)d_in[9];
    const float* vatt  = (const float*)d_in[10];
    const float* Wa    = (const float*)d_in[11];
    const float* Wfc   = (const float*)d_in[12];
    const float* bfc   = (const float*)d_in[13];
    float* out = (float*)d_out;

    // workspace carve-out (~49 MB total)
    char* p = (char*)d_ws;
    __bf16* WfcT  = (__bf16*)p; p += align256((size_t)VOCAB * UNITS * 2);
    __bf16* WkuT  = (__bf16*)p; p += align256((size_t)NZ * KXH * 2);
    float*  keys  = (float*)p;  p += align256((size_t)BATCH * TIN * UNITS * 4);
    __bf16* xh    = (__bf16*)p; p += align256((size_t)BATCH * KXH * 2);
    float*  z     = (float*)p;  p += align256((size_t)BATCH * NZ * 4);
    float*  h     = (float*)p;  p += align256((size_t)BATCH * UNITS * 4);
    float*  c     = (float*)p;  p += align256((size_t)BATCH * UNITS * 4);
    __bf16* attnA = (__bf16*)p; p += align256((size_t)BATCH * UNITS * 2);

    // one-time preprocessing
    {
        size_t n = (size_t)VOCAB * UNITS;
        k_wfcT<<<dim3((unsigned)((n + 255) / 256)), 256, 0, stream>>>(Wfc, WfcT);
    }
    {
        size_t n = (size_t)NZ * KXH;
        k_wkuT<<<dim3((unsigned)((n + 255) / 256)), 256, 0, stream>>>(Wk, Uk, WkuT);
    }
    k_keys<<<BATCH * TIN, 256, 0, stream>>>(mem, Wm, keys);
    {
        int n = BATCH * KXH;
        k_init<<<(n + 255) / 256, 256, 0, stream>>>(h0, c0, xh, h, c);
    }
    k_embed<<<BATCH, 128, 0, stream>>>(dec, emb, xh, 0);

    // sequential decode
    for (int t = 0; t < TOUT; ++t) {
        k_cell_gemm<<<dim3(16), 256, 0, stream>>>(xh, WkuT, bk, z);
        k_lstm<<<(BATCH * UNITS + 255) / 256, 256, 0, stream>>>(z, h, c, xh);
        k_attn<<<BATCH, 256, 0, stream>>>(h, keys, vatt, mem, Wq, Wa, attnA, xh);
        k_logits<<<dim3((NTILE_V + 7) / 8), 256, 0, stream>>>(attnA, WfcT, bfc, out, t);
        if (t + 1 < TOUT)
            k_embed<<<BATCH, 128, 0, stream>>>(dec, emb, xh, t + 1);
    }
}
